// FlexAttention_4741643895579
// MI455X (gfx1250) — matrix operand order repaired and checked by reading, not compiled, not run on hardware
//
#include <hip/hip_runtime.h>

// ---- CDNA5 WMMA vector types -------------------------------------------------
typedef __attribute__((ext_vector_type(16))) _Float16     v16h;
typedef __attribute__((ext_vector_type(8)))  float        v8f;
typedef __attribute__((ext_vector_type(4)))  unsigned int u32x4;
typedef __attribute__((ext_vector_type(2)))  unsigned int u32x2;
typedef __attribute__((ext_vector_type(4)))  _Float16     v4h;

#define S_LEN   2048
#define DHEAD   64
#define HEADS   16
#define QT      128          // q rows per block (8 waves x 16 rows)
#define KT      64           // kv rows per LDS tile
#define LSTR    72           // LDS row stride in halves (64 + 8 pad; 144B, 16B-aligned)

union FragU  { u32x4 u[2]; v16h h; };   // 32B fragment (8 VGPRs of halves)
union Pack4  { v4h   h;    u32x2 u; };  // 4 halves <-> 8 bytes

// ---- DPP16 lane moves (VALU pipe, keeps the DS unit free for WMMA operands) --
// ctrl: 0xB1 = quad_perm[1,0,3,2] (xor1), 0x4E = quad_perm[2,3,0,1] (xor2),
//       0x124 = row_ror:4, 0x128 = row_ror:8  (row = 16 lanes)
template <int CTRL>
__device__ __forceinline__ float dpp_mov(float x) {
    int xi = __float_as_int(x);
    int yi = __builtin_amdgcn_update_dpp(xi, xi, CTRL, 0xf, 0xf, true);
    return __int_as_float(yi);
}
__device__ __forceinline__ float row_max16(float t) {
    t = fmaxf(t, dpp_mov<0xB1>(t));    // + quad neighbor (xor1)
    t = fmaxf(t, dpp_mov<0x4E>(t));    // + quad pair (xor2) -> quad max
    t = fmaxf(t, dpp_mov<0x124>(t));   // + next quad (ror4)
    t = fmaxf(t, dpp_mov<0x128>(t));   // + remaining quads (ror8) -> row max
    return t;
}
__device__ __forceinline__ float row_sum16(float t) {
    t += dpp_mov<0xB1>(t);
    t += dpp_mov<0x4E>(t);
    t += dpp_mov<0x124>(t);
    t += dpp_mov<0x128>(t);
    return t;
}

__global__ __launch_bounds__(256)
void flex_attn_sink_kernel(const float* __restrict__ q,
                           const float* __restrict__ k,
                           const float* __restrict__ v,
                           const float* __restrict__ sinkw,
                           const int*   __restrict__ winp,
                           float*       __restrict__ out)
{
    __shared__ __align__(16) _Float16 Klds[KT * LSTR];        // K tile, row-major [kv][d]
    __shared__ __align__(16) _Float16 Vt  [DHEAD * LSTR];     // V tile, transposed [d][kv]
    __shared__ __align__(16) _Float16 Plds[8 * 16 * LSTR];    // per-wave P staging [16][64]

    const int tid  = threadIdx.x;
    const int wave = tid >> 5;
    const int lane = tid & 31;
    const int g    = lane >> 4;      // lane half-group (0/1)
    const int ln   = lane & 15;

    const int qtiles = S_LEN / QT;                  // 16
    const int bh = blockIdx.x / qtiles;             // b*H + h
    const int q0 = (blockIdx.x % qtiles) * QT;
    const int h  = bh % HEADS;
    const int W  = winp[0];

    // per-(b,h) base pointers so all inner addressing is 32-bit
    const float* kbh = k + (size_t)bh * S_LEN * DHEAD;
    const float* vbh = v + (size_t)bh * S_LEN * DHEAD;
    float*       obh = out + (size_t)bh * S_LEN * DHEAD;

    const int   qbase = q0 + wave * 16;             // this wave's 16 q rows
    const float scale = 0.125f;                     // 1/sqrt(64), folded into Q

    // ---- Q A-fragments (16x32 f16), loaded straight from global -------------
    // A layout: row M = lane&15; halves j=0..7 -> d = c*32 + g*8 + j,
    //           halves j=8..15 -> d = c*32 + g*8 + 16 + (j-8).
    const float* qrow = q + (size_t)bh * S_LEN * DHEAD + (qbase + ln) * DHEAD;
    v16h qf[2];
    #pragma unroll
    for (int c = 0; c < 2; ++c) {
        const float* qp = qrow + c * 32 + g * 8;
        #pragma unroll
        for (int j = 0; j < 8; ++j) {
            qf[c][j]     = (_Float16)(qp[j]      * scale);
            qf[c][j + 8] = (_Float16)(qp[j + 16] * scale);
        }
    }

    // ---- online-softmax state: sink column == extra logit with zero value ---
    const float sink = sinkw[h];
    float m_i[8], l_i[8];
    v8f acc[4] = { {}, {}, {}, {} };                // 16x64 output accumulator
    #pragma unroll
    for (int r = 0; r < 8; ++r) { m_i[r] = sink; l_i[r] = 1.0f; }

    int kv_start = q0 - W;
    if (kv_start < 0) kv_start = 0;
    kv_start &= ~(KT - 1);
    const int kv_end = q0 + QT;                     // kv <= q < q0+QT <= S, 64-aligned

    _Float16* Pw = Plds + wave * 16 * LSTR;

    for (int kv0 = kv_start; kv0 < kv_end; kv0 += KT) {
        const bool pref = (kv0 + KT < kv_end);      // wave-uniform prefetch predicate

        // ---- cooperative staging: K row-major f16, V transposed f16 ---------
        // All addresses provably in-bounds: kv0+63 <= kv_end-1 <= S-1.
        #pragma unroll
        for (int i = 0; i < 4; ++i) {
            const int f4   = tid + i * 256;         // 0..1023 float4 slots
            const int row  = f4 >> 4;               // kv row within tile
            const int col4 = f4 & 15;               // float4 column in d
            const int off  = (kv0 + row) * DHEAD + col4 * 4;
            float4 kd = *(const float4*)(kbh + off);
            float4 vd = *(const float4*)(vbh + off);
            if (pref) {                             // next tile into L2
                __builtin_prefetch(kbh + off + KT * DHEAD, 0, 1);
                __builtin_prefetch(vbh + off + KT * DHEAD, 0, 1);
            }
            Pack4 pk;
            pk.h[0] = (_Float16)kd.x; pk.h[1] = (_Float16)kd.y;
            pk.h[2] = (_Float16)kd.z; pk.h[3] = (_Float16)kd.w;
            *(u32x2*)&Klds[row * LSTR + col4 * 4] = pk.u;
            const int d0 = col4 * 4;
            Vt[(d0 + 0) * LSTR + row] = (_Float16)vd.x;
            Vt[(d0 + 1) * LSTR + row] = (_Float16)vd.y;
            Vt[(d0 + 2) * LSTR + row] = (_Float16)vd.z;
            Vt[(d0 + 3) * LSTR + row] = (_Float16)vd.w;
        }
        __syncthreads();

        // wave-uniform predicates: any element visible / all elements visible
        const bool active = (kv0 <= qbase + 15) && (kv0 + KT - 1 >= qbase - W);
        const bool full   = (kv0 + KT - 1 <= qbase) && (qbase + 15 - kv0 <= W);

        if (active) {
            // ---- S = Q K^T : 4 n-tiles x 2 k-chunks of wmma_f32_16x16x32_f16
            v8f st[4];
            #pragma unroll
            for (int n = 0; n < 4; ++n) {
                v8f c = {};
                #pragma unroll
                for (int cc = 0; cc < 2; ++cc) {
                    // B layout: col N = ln, d runs [cc*32+g*8, +8) and [cc*32+g*8+16, +8)
                    const _Float16* kp = &Klds[(n * 16 + ln) * LSTR + cc * 32 + g * 8];
                    FragU kb;
                    kb.u[0] = *(const u32x4*)kp;
                    kb.u[1] = *(const u32x4*)(kp + 16);
                    c = __builtin_amdgcn_wmma_f32_16x16x32_f16(
                            false, qf[cc], false, kb.h, (short)0, c, false, false);
                }
                st[n] = c;
            }

            // ---- masking: only boundary tiles pay for it --------------------
            if (!full) {
                #pragma unroll
                for (int r = 0; r < 8; ++r) {
                    const int qi = qbase + r + g * 8;
                    const int hi = qi - kv0;        // keep: col <= hi
                    const int lo = hi - W;          // keep: col >= lo
                    #pragma unroll
                    for (int n = 0; n < 4; ++n) {
                        const int col = n * 16 + ln;
                        if (col > hi || col < lo) st[n][r] = -1.0e30f;
                    }
                }
            }

            // ---- online softmax update (C layout: row = r + g*8) -----------
            #pragma unroll
            for (int r = 0; r < 8; ++r) {
                float t = fmaxf(fmaxf(st[0][r], st[1][r]),
                                fmaxf(st[2][r], st[3][r]));
                t = row_max16(t);
                const float mn = fmaxf(m_i[r], t);
                const float sc = __expf(m_i[r] - mn);
                m_i[r] = mn;
                l_i[r] *= sc;
                #pragma unroll
                for (int n = 0; n < 4; ++n) acc[n][r] *= sc;
                float ps = 0.0f;
                #pragma unroll
                for (int n = 0; n < 4; ++n) {
                    const float p = __expf(st[n][r] - mn);
                    ps += p;
                    Pw[(r + g * 8) * LSTR + n * 16 + ln] = (_Float16)p;
                }
                l_i[r] += row_sum16(ps);
            }

            // ---- O += P V : re-gather P as A-fragments from per-wave LDS ---
            #pragma unroll
            for (int cc = 0; cc < 2; ++cc) {
                const _Float16* pp = &Pw[ln * LSTR + cc * 32 + g * 8];
                FragU pa;
                pa.u[0] = *(const u32x4*)pp;         // kv run [base, base+8)
                pa.u[1] = *(const u32x4*)(pp + 16);  // kv run [base+16, base+24)
                #pragma unroll
                for (int n = 0; n < 4; ++n) {
                    // B layout from Vt: col d = n*16+ln, kv runs [base, base+8) and [base+16, base+24)
                    const _Float16* vp2 = &Vt[(n * 16 + ln) * LSTR + cc * 32 + g * 8];
                    FragU vb;
                    vb.u[0] = *(const u32x4*)vp2;
                    vb.u[1] = *(const u32x4*)(vp2 + 16);
                    acc[n] = __builtin_amdgcn_wmma_f32_16x16x32_f16(
                                 false, pa.h, false, vb.h, (short)0, acc[n], false, false);
                }
            }
        }
        __syncthreads();
    }

    // ---- epilogue: out = acc / l ------------------------------------------
    #pragma unroll
    for (int r = 0; r < 8; ++r) {
        const int qi  = qbase + r + g * 8;
        const float inv = __builtin_amdgcn_rcpf(l_i[r]);
        float* op = obh + qi * DHEAD;
        #pragma unroll
        for (int n = 0; n < 4; ++n)
            op[n * 16 + ln] = acc[n][r] * inv;
    }
}

extern "C" void kernel_launch(void* const* d_in, const int* in_sizes, int n_in,
                              void* d_out, int out_size, void* d_ws, size_t ws_size,
                              hipStream_t stream) {
    (void)in_sizes; (void)n_in; (void)out_size; (void)d_ws; (void)ws_size;
    const float* q     = (const float*)d_in[0];
    const float* k     = (const float*)d_in[1];
    const float* v     = (const float*)d_in[2];
    const float* sinkw = (const float*)d_in[3];
    const int*   win   = (const int*)d_in[4];
    float* out = (float*)d_out;

    // B*H * (S/QT) = 2*16 * 16 = 512 blocks, 256 threads (8 wave32)
    dim3 grid(2 * HEADS * (S_LEN / QT));
    flex_attn_sink_kernel<<<grid, 256, 0, stream>>>(q, k, v, sinkw, win, out);
}
